// AMGSGAN_50878182588759
// MI455X (gfx1250) — matrix operand order repaired and checked by reading, not compiled, not run on hardware
//
#include <hip/hip_runtime.h>
#include <hip/hip_bf16.h>
#include <math.h>
#include <stdint.h>

typedef __attribute__((ext_vector_type(16))) _Float16 v16h;
typedef __attribute__((ext_vector_type(8)))  _Float16 v8h;
typedef __attribute__((ext_vector_type(8)))  float    v8f;

#define FDIM 128
#define NBANDS 31

// ---------------------------------------------------------------------------
// Scalar 3x3 encoder conv (pad=1, dil=1) + ReLU. f32 [Cin][H][W] -> f16 [H*W][128]
// ---------------------------------------------------------------------------
__global__ void k_enc(const float* __restrict__ in, const float* __restrict__ w,
                      const float* __restrict__ bias, _Float16* __restrict__ out,
                      int H, int wShift, int Cin) {
  const int W = 1 << wShift;
  const int HW = H << wShift;
  int tid = blockIdx.x * blockDim.x + threadIdx.x;
  if (tid >= HW * FDIM) return;
  int oc  = tid & (FDIM - 1);
  int pid = tid >> 7;
  int y = pid >> wShift, x = pid & (W - 1);
  float acc = bias[oc];
  for (int ky = 0; ky < 3; ++ky) {
    int sy = y + ky - 1; if ((unsigned)sy >= (unsigned)H) continue;
    for (int kx = 0; kx < 3; ++kx) {
      int sx = x + kx - 1; if ((unsigned)sx >= (unsigned)W) continue;
      const float* ip = in + ((sy << wShift) + sx);
      const float* wp = w + (oc * Cin) * 9 + ky * 3 + kx;
      for (int ic = 0; ic < Cin; ++ic)
        acc += ip[ic * HW] * wp[ic * 9];
    }
  }
  acc = acc > 0.f ? acc : 0.f;
  out[pid * FDIM + oc] = (_Float16)acc;
}

// ---------------------------------------------------------------------------
// Conv weight prep: f32 [oc][ic=128][3][3] -> f16 [oc][tap=9][ic=128]
// ---------------------------------------------------------------------------
__global__ void k_prep_cw(const float* __restrict__ w, _Float16* __restrict__ out) {
  int tid = blockIdx.x * blockDim.x + threadIdx.x;
  if (tid >= FDIM * 9 * FDIM) return;
  int ic  = tid & (FDIM - 1);
  int tap = (tid >> 7) % 9;
  int oc  = tid / (FDIM * 9);
  out[tid] = (_Float16)w[(oc * FDIM + ic) * 9 + tap];
}

// ---------------------------------------------------------------------------
// MLP weight prep: f32 [K][N] -> f16 [N][Kpad] (transpose + zero pad)
// ---------------------------------------------------------------------------
__global__ void k_prep_wt(const float* __restrict__ w, _Float16* __restrict__ out,
                          int K, int N, int Kpad) {
  int tid = blockIdx.x * blockDim.x + threadIdx.x;
  if (tid >= N * Kpad) return;
  int kp = tid % Kpad;
  int n  = tid / Kpad;
  out[tid] = (kp < K) ? (_Float16)w[kp * N + n] : (_Float16)0.0f;
}

// ---------------------------------------------------------------------------
// WMMA implicit-GEMM 3x3 dilated conv, 128->128, no ReLU.
// Block = 256 threads = 8 waves; blockIdx.x covers 32 consecutive pixels
// (two 16-pixel tiles). Each wave owns one 16-out-channel tile with dual
// accumulators so every B (weight) tile is loaded once, used twice.
// ---------------------------------------------------------------------------
__global__ void k_conv_wmma(const _Float16* __restrict__ in,
                            const _Float16* __restrict__ wq,
                            const float* __restrict__ bias,
                            _Float16* __restrict__ out,
                            int H, int wShift, int dil) {
  const int W = 1 << wShift;
  const int wave = threadIdx.x >> 5;
  const int lane = threadIdx.x & 31;
  const int r  = lane & 15;   // A: pixel row / B: out-channel col
  const int hh = lane >> 4;   // lane half
  const int oc0  = wave * 16;
  const int pid0 = blockIdx.x * 32;
  const int pA0 = pid0 + r, pA1 = pid0 + 16 + r;
  const int y0 = pA0 >> wShift, x0 = pA0 & (W - 1);
  const int y1 = pA1 >> wShift, x1 = pA1 & (W - 1);

  v8f acc0 = {}, acc1 = {};
  for (int ky = 0; ky < 3; ++ky) {
    for (int kx = 0; kx < 3; ++kx) {
      const int tap = ky * 3 + kx;
      const int dy = (ky - 1) * dil, dx = (kx - 1) * dil;
      const int sy0 = y0 + dy, sx0 = x0 + dx;
      const int sy1 = y1 + dy, sx1 = x1 + dx;
      const bool inb0 = (unsigned)sy0 < (unsigned)H && (unsigned)sx0 < (unsigned)W;
      const bool inb1 = (unsigned)sy1 < (unsigned)H && (unsigned)sx1 < (unsigned)W;
      const _Float16* ap0 = in + (((sy0 << wShift) + sx0) * FDIM);
      const _Float16* ap1 = in + (((sy1 << wShift) + sx1) * FDIM);
      const _Float16* bp  = wq + ((oc0 + r) * 9 + tap) * FDIM + hh * 16;
      if (tap < 8) __builtin_prefetch(bp + FDIM, 0, 3);    // next tap, WGP scope
      for (int kt = 0; kt < 4; ++kt) {
        v16h bv = *(const v16h*)(bp + kt * 32);            // B: k = h*16 + e
        v8h lo0 = {}, hi0 = {}, lo1 = {}, hi1 = {};
        if (inb0) {
          lo0 = *(const v8h*)(ap0 + kt * 32 + hh * 16);     // k = h*16 + e
          hi0 = *(const v8h*)(ap0 + kt * 32 + hh * 16 + 8);// k = h*16 + 8 + e
        }
        if (inb1) {
          lo1 = *(const v8h*)(ap1 + kt * 32 + hh * 16);
          hi1 = *(const v8h*)(ap1 + kt * 32 + hh * 16 + 8);
        }
        v16h a0, a1;
        for (int e = 0; e < 8; ++e) {
          a0[e] = lo0[e]; a0[e + 8] = hi0[e];
          a1[e] = lo1[e]; a1[e + 8] = hi1[e];
        }
        acc0 = __builtin_amdgcn_wmma_f32_16x16x32_f16(false, a0, false, bv,
                                                      (short)0, acc0, false, false);
        acc1 = __builtin_amdgcn_wmma_f32_16x16x32_f16(false, a1, false, bv,
                                                      (short)0, acc1, false, false);
      }
    }
  }
  const float bc = bias[oc0 + r];
  for (int v = 0; v < 8; ++v) {
    int prow0 = pid0 + v + 8 * hh;                         // D row M = v + 8*h
    int prow1 = pid0 + 16 + v + 8 * hh;
    out[prow0 * FDIM + oc0 + r] = (_Float16)(acc0[v] + bc);
    out[prow1 * FDIM + oc0 + r] = (_Float16)(acc1[v] + bc);
  }
}

// ---------------------------------------------------------------------------
// MLP WMMA GEMM helpers (32 rows = two 16-row tiles, dual accumulators).
// ---------------------------------------------------------------------------
__device__ __forceinline__ void gemm_f16(const _Float16* sIn, int inStride,
                                         const _Float16* Wt, int Kpad,
                                         const float* bias, int N,
                                         _Float16* sOut, int outStride,
                                         bool relu, int tid) {
  const int wave = tid >> 5, lane = tid & 31;
  const int r = lane & 15, hh = lane >> 4;
  const int ktiles = Kpad >> 5, ctiles = N >> 4;
  for (int ct = wave; ct < ctiles; ct += 8) {
    v8f acc0 = {}, acc1 = {};
    for (int kt = 0; kt < ktiles; ++kt) {
      const _Float16* bp = Wt + (ct * 16 + r) * Kpad + kt * 32 + hh * 16;
      v16h bv = *(const v16h*)bp;
      if (kt + 1 < ktiles) __builtin_prefetch(bp + 32, 0, 3);  // WGP scope
      const _Float16* ap0 = sIn + r * inStride + kt * 32 + hh * 16;
      const _Float16* ap1 = ap0 + 16 * inStride;
      v8h lo0 = *(const v8h*)ap0, hi0 = *(const v8h*)(ap0 + 8);
      v8h lo1 = *(const v8h*)ap1, hi1 = *(const v8h*)(ap1 + 8);
      v16h a0, a1;
      for (int e = 0; e < 8; ++e) {
        a0[e] = lo0[e]; a0[e + 8] = hi0[e];
        a1[e] = lo1[e]; a1[e + 8] = hi1[e];
      }
      acc0 = __builtin_amdgcn_wmma_f32_16x16x32_f16(false, a0, false, bv,
                                                    (short)0, acc0, false, false);
      acc1 = __builtin_amdgcn_wmma_f32_16x16x32_f16(false, a1, false, bv,
                                                    (short)0, acc1, false, false);
    }
    const float bc = bias[ct * 16 + r];
    for (int v = 0; v < 8; ++v) {
      float v0 = acc0[v] + bc, v1 = acc1[v] + bc;
      if (relu) { v0 = v0 > 0.f ? v0 : 0.f; v1 = v1 > 0.f ? v1 : 0.f; }
      sOut[(v + 8 * hh) * outStride + ct * 16 + r]        = (_Float16)v0;
      sOut[(16 + v + 8 * hh) * outStride + ct * 16 + r]   = (_Float16)v1;
    }
  }
}

__device__ __forceinline__ void gemm_f32(const _Float16* sIn, int inStride,
                                         const _Float16* Wt, int Kpad,
                                         const float* bias, int N,
                                         float* sOut, int outStride, int tid) {
  const int wave = tid >> 5, lane = tid & 31;
  const int r = lane & 15, hh = lane >> 4;
  const int ktiles = Kpad >> 5, ctiles = N >> 4;
  for (int ct = wave; ct < ctiles; ct += 8) {
    v8f acc0 = {}, acc1 = {};
    for (int kt = 0; kt < ktiles; ++kt) {
      const _Float16* bp = Wt + (ct * 16 + r) * Kpad + kt * 32 + hh * 16;
      v16h bv = *(const v16h*)bp;
      const _Float16* ap0 = sIn + r * inStride + kt * 32 + hh * 16;
      const _Float16* ap1 = ap0 + 16 * inStride;
      v8h lo0 = *(const v8h*)ap0, hi0 = *(const v8h*)(ap0 + 8);
      v8h lo1 = *(const v8h*)ap1, hi1 = *(const v8h*)(ap1 + 8);
      v16h a0, a1;
      for (int e = 0; e < 8; ++e) {
        a0[e] = lo0[e]; a0[e + 8] = hi0[e];
        a1[e] = lo1[e]; a1[e + 8] = hi1[e];
      }
      acc0 = __builtin_amdgcn_wmma_f32_16x16x32_f16(false, a0, false, bv,
                                                    (short)0, acc0, false, false);
      acc1 = __builtin_amdgcn_wmma_f32_16x16x32_f16(false, a1, false, bv,
                                                    (short)0, acc1, false, false);
    }
    const float bc = bias[ct * 16 + r];
    for (int v = 0; v < 8; ++v) {
      sOut[(v + 8 * hh) * outStride + ct * 16 + r]      = acc0[v] + bc;
      sOut[(16 + v + 8 * hh) * outStride + ct * 16 + r] = acc1[v] + bc;
    }
  }
}

// ---------------------------------------------------------------------------
// MLP megakernel for one scale. Block = 8 HR pixels -> 32 rows (4 offsets x 8).
// Guide half of the input rows is copied global->LDS with async-to-LDS ops
// (ASYNCcnt); feature gather + rel is filled by VALU while that is in flight.
// Then three WMMA GEMM layers LDS<->LDS and a per-pixel softmax epilogue.
// ---------------------------------------------------------------------------
__global__ void k_mlp(const _Float16* __restrict__ Hf,  // LR feat [64*64][128]
                      const _Float16* __restrict__ Mg,  // HR guide [256*256][128]
                      const _Float16* __restrict__ Wt1, const float* __restrict__ b1,
                      const _Float16* __restrict__ Wt2, const float* __restrict__ b2,
                      const _Float16* __restrict__ Wt3, const float* __restrict__ b3,
                      const float* __restrict__ coeffp, int accum,
                      float* __restrict__ outp) {
  __shared__ _Float16 sA[32 * 288];
  __shared__ _Float16 sB[32 * 256];
  const int tid = threadIdx.x;
  const int blk = blockIdx.x;

  // ---- stage 0a: async global->LDS copy of guide rows (always valid) ----
  {
    const int row   = tid >> 3;          // 0..31
    const int pr    = row & 7;
    const int pid   = blk * 8 + pr;
    const int chunk = (tid & 7) * 2;     // two 16B chunks per thread
    _Float16* dst = sA + row * 288 + 128 + chunk * 8;
    const _Float16* src = Mg + pid * FDIM + chunk * 8;
    unsigned d0 = (unsigned)(uintptr_t)dst;          // low 32 bits = LDS offset
    unsigned long long s0 = (unsigned long long)(uintptr_t)src;
    asm volatile("global_load_async_to_lds_b128 %0, %1, off"
                 :: "v"(d0), "v"(s0) : "memory");
    unsigned d1 = d0 + 16;
    unsigned long long s1 = s0 + 16;
    asm volatile("global_load_async_to_lds_b128 %0, %1, off"
                 :: "v"(d1), "v"(s1) : "memory");
  }

  // ---- stage 0b: feature gather + rel + zero padding (cols 0..127, 256..287)
  {
    const int row  = tid >> 3;        // 0..31
    const int part = tid & 7;         // 8 threads per row, 20 cols each
    const int t  = row >> 3;          // offset index 0..3
    const int pr = row & 7;           // pixel within block
    const int pid = blk * 8 + pr;
    const int i = pid >> 8, j = pid & 255;
    const float cy = (2 * i + 1) * (1.0f / 256.0f) - 1.0f;
    const float cx = (2 * j + 1) * (1.0f / 256.0f) - 1.0f;
    const float oy = (t & 2) ? 1.0f : -1.0f;
    const float ox = (t & 1) ? 1.0f : -1.0f;
    const float fy = (cy + oy * (1.0f / 64.0f) + 1.0f) * 32.0f - 0.5f;
    const float fx = (cx + ox * (1.0f / 64.0f) + 1.0f) * 32.0f - 0.5f;
    const int iy = (int)floorf(fy + 0.5f);
    const int ix = (int)floorf(fx + 0.5f);
    const bool valid = (iy >= 0 && iy < 64 && ix >= 0 && ix < 64);
    const int iyc = iy < 0 ? 0 : (iy > 63 ? 63 : iy);
    const int ixc = ix < 0 ? 0 : (ix > 63 ? 63 : ix);
    const float qy = valid ? ((2 * iyc + 1) * (1.0f / 64.0f) - 1.0f) : 0.0f;
    const float qx = valid ? ((2 * ixc + 1) * (1.0f / 64.0f) - 1.0f) : 0.0f;
    const float rely = (cy - qy) * 64.0f;
    const float relx = (cx - qx) * 64.0f;
    const _Float16* fp = Hf + (iyc * 64 + ixc) * FDIM;
    _Float16* arow = sA + row * 288;
    for (int cc = part * 20; cc < part * 20 + 20; ++cc) {
      const int c = (cc < 128) ? cc : cc + 128;  // 128..159 -> 256..287
      _Float16 vv;
      if (c < 128)        vv = valid ? fp[c] : (_Float16)0.0f;
      else if (c == 256)  vv = (_Float16)rely;
      else if (c == 257)  vv = (_Float16)relx;
      else                vv = (_Float16)0.0f;
      arow[c] = vv;
    }
  }
  asm volatile("s_wait_asynccnt 0x0" ::: "memory");
  __syncthreads();
  // ---- layer 1: 32x288 @ 288x256 -> 32x256, ReLU ----
  gemm_f16(sA, 288, Wt1, 288, b1, 256, sB, 256, true, tid);
  __syncthreads();
  // ---- layer 2: 32x256 @ 256x128 -> 32x128, ReLU (reuse sA) ----
  gemm_f16(sB, 256, Wt2, 256, b2, 128, sA, 128, true, tid);
  __syncthreads();
  // ---- layer 3: 32x128 @ 128x32 -> 32x32 f32 (reuse sB as f32) ----
  gemm_f32(sA, 128, Wt3, 128, b3, 32, (float*)sB, 32, tid);
  __syncthreads();
  // ---- softmax over 4 offsets + band combine ----
  if (tid < 8 * NBANDS) {
    const float* so = (const float*)sB;
    const int pr = tid / NBANDS, band = tid % NBANDS;
    float l0 = so[(0 * 8 + pr) * 32 + 31];
    float l1 = so[(1 * 8 + pr) * 32 + 31];
    float l2 = so[(2 * 8 + pr) * 32 + 31];
    float l3 = so[(3 * 8 + pr) * 32 + 31];
    float m = fmaxf(fmaxf(l0, l1), fmaxf(l2, l3));
    float e0 = __expf(l0 - m), e1 = __expf(l1 - m);
    float e2 = __expf(l2 - m), e3 = __expf(l3 - m);
    float inv = 1.0f / (e0 + e1 + e2 + e3);
    float val = so[(0 * 8 + pr) * 32 + band] * e0 +
                so[(1 * 8 + pr) * 32 + band] * e1 +
                so[(2 * 8 + pr) * 32 + band] * e2 +
                so[(3 * 8 + pr) * 32 + band] * e3;
    val *= inv;
    const float coeff = *coeffp;
    const int idx = band * 65536 + blk * 8 + pr;
    if (accum) outp[idx] += coeff * val;
    else       outp[idx]  = coeff * val;
  }
}

// ---------------------------------------------------------------------------
extern "C" void kernel_launch(void* const* d_in, const int* in_sizes, int n_in,
                              void* d_out, int out_size, void* d_ws, size_t ws_size,
                              hipStream_t stream) {
  (void)in_sizes; (void)n_in; (void)out_size; (void)ws_size;
  const float* HSI   = (const float*)d_in[0];
  const float* MSI   = (const float*)d_in[1];
  const float* ehw   = (const float*)d_in[2];
  const float* ehb   = (const float*)d_in[3];
  const float* emw   = (const float*)d_in[4];
  const float* emb   = (const float*)d_in[5];
  const float* cw_f32[6] = {(const float*)d_in[6],  (const float*)d_in[8],
                            (const float*)d_in[10], (const float*)d_in[12],
                            (const float*)d_in[14], (const float*)d_in[16]};
  const float* cb[6]     = {(const float*)d_in[7],  (const float*)d_in[9],
                            (const float*)d_in[11], (const float*)d_in[13],
                            (const float*)d_in[15], (const float*)d_in[17]};
  const float* w1 = (const float*)d_in[18];  const float* bb1 = (const float*)d_in[19];
  const float* w2 = (const float*)d_in[20];  const float* bb2 = (const float*)d_in[21];
  const float* w3 = (const float*)d_in[22];  const float* bb3 = (const float*)d_in[23];
  const float* coeffs[3] = {(const float*)d_in[24], (const float*)d_in[25],
                            (const float*)d_in[26]};
  float* out = (float*)d_out;

  // workspace carve (256-byte aligned regions)
  size_t off = 0;
  char* base = (char*)d_ws;
  auto carve = [&](size_t bytes) -> void* {
    void* p = base + off;
    off += (bytes + 255) & ~(size_t)255;
    return p;
  };
  _Float16* hsiE = (_Float16*)carve((size_t)64 * 64 * 128 * 2);
  _Float16* msiE = (_Float16*)carve((size_t)256 * 256 * 128 * 2);
  _Float16* Hf[3], *Mf[3], *cw[6];
  for (int s = 0; s < 3; ++s) Hf[s] = (_Float16*)carve((size_t)64 * 64 * 128 * 2);
  for (int s = 0; s < 3; ++s) Mf[s] = (_Float16*)carve((size_t)256 * 256 * 128 * 2);
  for (int s = 0; s < 6; ++s) cw[s] = (_Float16*)carve((size_t)128 * 9 * 128 * 2);
  _Float16* Wt1 = (_Float16*)carve((size_t)256 * 288 * 2);
  _Float16* Wt2 = (_Float16*)carve((size_t)128 * 256 * 2);
  _Float16* Wt3 = (_Float16*)carve((size_t)32 * 128 * 2);

  // encoders (scalar, small)
  k_enc<<<(64 * 64 * 128) / 256, 256, 0, stream>>>(HSI, ehw, ehb, hsiE, 64, 6, 31);
  k_enc<<<(256 * 256 * 128) / 256, 256, 0, stream>>>(MSI, emw, emb, msiE, 256, 8, 4);

  // weight prep
  for (int s = 0; s < 6; ++s)
    k_prep_cw<<<(128 * 9 * 128) / 256, 256, 0, stream>>>(cw_f32[s], cw[s]);
  k_prep_wt<<<(256 * 288) / 256, 256, 0, stream>>>(w1, Wt1, 258, 256, 288);
  k_prep_wt<<<(128 * 256) / 256, 256, 0, stream>>>(w2, Wt2, 256, 128, 256);
  k_prep_wt<<<(32 * 128) / 256, 256, 0, stream>>>(w3, Wt3, 128, 32, 128);

  // dilated 128->128 convs via WMMA implicit GEMM (32 pixels / block)
  for (int s = 0; s < 3; ++s)
    k_conv_wmma<<<(64 * 64) / 32, 256, 0, stream>>>(hsiE, cw[s], cb[s], Hf[s],
                                                    64, 6, s + 1);
  for (int s = 0; s < 3; ++s)
    k_conv_wmma<<<(256 * 256) / 32, 256, 0, stream>>>(msiE, cw[3 + s], cb[3 + s],
                                                      Mf[s], 256, 8, s + 1);

  // per-scale MLP + softmax combine into output (scale 0 writes, 1/2 accumulate)
  for (int s = 0; s < 3; ++s)
    k_mlp<<<(256 * 256) / 8, 256, 0, stream>>>(Hf[s], Mf[s], Wt1, bb1, Wt2, bb2,
                                               Wt3, bb3, coeffs[s], s == 0 ? 0 : 1,
                                               out);
}
